// ModularPolicyGRU_29850022707286
// MI455X (gfx1250) — matrix operand order repaired and checked by reading, not compiled, not run on hardware
//
#include <hip/hip_runtime.h>
#include <hip/hip_bf16.h>

typedef __bf16 bf16;
typedef __attribute__((ext_vector_type(16))) __bf16 v16bf;
typedef __attribute__((ext_vector_type(8)))  __bf16 v8bf;
typedef __attribute__((ext_vector_type(8)))  float  v8f;
typedef int v4i __attribute__((vector_size(16)));   // matches builtin param type

#define GAS __attribute__((address_space(1)))
#define LAS __attribute__((address_space(3)))

#if defined(__HIP_DEVICE_COMPILE__) && __has_builtin(__builtin_amdgcn_global_load_async_to_lds_b128)
#define HAVE_ASYNC_LDS 1
#else
#define HAVE_ASYNC_LDS 0
#endif

#if defined(__HIP_DEVICE_COMPILE__) && __has_builtin(__builtin_amdgcn_s_wait_asynccnt)
#define WAIT_ASYNC() __builtin_amdgcn_s_wait_asynccnt(0)
#elif HAVE_ASYNC_LDS
#define WAIT_ASYNC() asm volatile("s_wait_asynccnt 0x0" ::: "memory")
#else
#define WAIT_ASYNC() ((void)0)
#endif

constexpr int kB  = 1024;   // batch
constexpr int kH  = 2048;   // hidden
constexpr int kI  = 128;    // input size
constexpr int kKA = 2176;   // I + H
constexpr int kO  = 64;     // output size
constexpr int kMS = 512;    // module size
constexpr int kLdsPitch = 80;  // halfs per LDS row (160B: 16B-aligned, bank-spread)

// CONN_DELAY flattened [target][source]
__constant__ int c_delay[16] = {0,1,2,1, 1,0,1,2, 2,1,0,1, 1,2,1,0};

// ---------------- prep kernels ----------------

__global__ void k_f32_to_bf16(const float* __restrict__ s, bf16* __restrict__ d, int n) {
    int i = blockIdx.x * blockDim.x + threadIdx.x;
    if (i < n) d[i] = (bf16)s[i];
}

// Y padded to [128][2048] (rows >= 64 are zero) so the output GEMM needs no bounds checks
__global__ void k_prep_y(const float* __restrict__ Y, bf16* __restrict__ Yp) {
    int i = blockIdx.x * blockDim.x + threadIdx.x;   // 128*2048
    if (i >= 128 * kH) return;
    int row = i >> 11, col = i & (kH - 1);
    Yp[i] = (row < kO) ? (bf16)Y[row * kH + col] : (bf16)0.0f;
}

// Build bf16 x and the 3 delayed hidden copies:
// Hd[0]=h_prev, Hd[1]=h_buffer slot0, Hd[2]=h_buffer slot1  (buffer shift semantics)
__global__ void k_prep_inputs(const float* __restrict__ x, const float* __restrict__ h_prev,
                              const float* __restrict__ h_buffer,
                              bf16* __restrict__ Xbf, bf16* __restrict__ Hd) {
    int i = blockIdx.x * blockDim.x + threadIdx.x;   // B*H
    if (i >= kB * kH) return;
    if (i < kB * kI) Xbf[i] = (bf16)x[i];
    Hd[i]               = (bf16)h_prev[i];
    Hd[kB * kH + i]     = (bf16)h_buffer[(size_t)i * 3 + 0];
    Hd[2 * kB * kH + i] = (bf16)h_buffer[(size_t)i * 3 + 1];
}

// ---------------- fused WMMA GEMM ----------------
// MODE 0: Z  = sigmoid(A_i @ Wz[sl]^T + bz)           -> Zf (f32)
// MODE 1: R  = sigmoid(A_i @ Wr^T + br); Rh = R*h_del -> Rh (bf16)
// MODE 2: Ht = tanh(Ah_i @ Wh[sl]^T + bh); h_new = (1-z)*h_prev + z*Ht -> outF + Hn(bf16)
// MODE 3: y  = sigmoid(Hn @ Ypad^T + bY)              -> outF
template<int MODE>
__global__ __launch_bounds__(256)
void gemm_wmma(const bf16* __restrict__ Xbf,
               const bf16* __restrict__ Hd,   // [3][B][H]
               const bf16* __restrict__ Rh,   // [4][B][H]
               const bf16* __restrict__ Hn,   // [B][H]
               const bf16* __restrict__ W,    // bf16 weights, row-major [Mw][ldw]
               int ldw, int Ktot,
               const float* __restrict__ bias,
               const float* __restrict__ Zf,      // MODE 2
               const float* __restrict__ h_prev,  // MODE 2
               float* __restrict__ outF,
               bf16*  __restrict__ outB) {
    // double-buffered stage tiles: 128 rows x 64 K (halfs), 16B-aligned rows
    __shared__ __align__(16) bf16 As[2][128][kLdsPitch];
    __shared__ __align__(16) bf16 Bs[2][128][kLdsPitch];

    const int tid   = threadIdx.x;
    const int lane  = tid & 31;
    const int wave  = tid >> 5;
    const int waveM = wave & 1;      // 2 groups of 64 batch rows
    const int waveN = wave >> 1;     // 4 groups of 32 out units
    const int lm    = lane & 15;
    const int lh    = lane >> 4;
    const int mTile = blockIdx.y * 128;
    const int nTile = blockIdx.x * 128;
    const int mod   = blockIdx.z;

    const int nWBase = nTile + ((MODE == 0 || MODE == 2) ? mod * kMS : 0);
    const bf16* wBase = W + (size_t)nWBase * ldw;

    // Resolve A-matrix source for a 64-wide K tile (all segment boundaries are 64-aligned)
    auto resolveA = [&](int kt, const bf16*& asrc, int& lda) {
        if (MODE == 3)    { asrc = Hn + kt; lda = kH; }
        else if (kt < kI) { asrc = Xbf + kt; lda = kI; }
        else {
            int j0 = kt - kI;
            if (MODE == 2) { asrc = Rh + (size_t)mod * kB * kH + j0; lda = kH; }
            else           { int d = c_delay[mod * 4 + (j0 >> 9)];
                             asrc = Hd + (size_t)d * kB * kH + j0; lda = kH; }
        }
    };

    // Stage one 128x64 A tile + 128x64 B tile into buffer `buf`.
    auto stageTile = [&](int kt, int buf) {
        const bf16* asrc; int lda;
        resolveA(kt, asrc, lda);
        const bf16* bsrc = wBase + kt;
        #pragma unroll
        for (int p = 0; p < 4; ++p) {
            int idx = p * 256 + tid;
            int row = idx >> 3;            // 0..127
            int cg  = (idx & 7) * 8;       // 0..56 halfs (16B chunks)
#if HAVE_ASYNC_LDS
            __builtin_amdgcn_global_load_async_to_lds_b128(
                (GAS v4i*)(asrc + (size_t)(mTile + row) * lda + cg),
                (LAS v4i*)&As[buf][row][cg], 0, 0);
            __builtin_amdgcn_global_load_async_to_lds_b128(
                (GAS v4i*)(bsrc + (size_t)row * ldw + cg),
                (LAS v4i*)&Bs[buf][row][cg], 0, 0);
#else
            *(uint4*)&As[buf][row][cg] = *(const uint4*)&asrc[(size_t)(mTile + row) * lda + cg];
            *(uint4*)&Bs[buf][row][cg] = *(const uint4*)&bsrc[(size_t)row * ldw + cg];
#endif
        }
    };

    v8f acc[4][2];
    #pragma unroll
    for (int f = 0; f < 4; ++f)
        #pragma unroll
        for (int g = 0; g < 2; ++g)
            acc[f][g] = (v8f){0.f,0.f,0.f,0.f,0.f,0.f,0.f,0.f};

    stageTile(0, 0);
    int buf = 0;
    for (int kt = 0; kt < Ktot; kt += 64) {
        WAIT_ASYNC();        // my async copies into As/Bs[buf] are complete
        __syncthreads();     // everyone's copies complete; buf^1 no longer being read

        if (kt + 64 < Ktot) stageTile(kt + 64, buf ^ 1);            // overlap with compute
        if (kt + 128 < Ktot)                                         // L2 warm for stage+2
            __builtin_prefetch(wBase + kt + 128 + (size_t)(tid >> 1) * ldw, 0, 1);

        #pragma unroll
        for (int kk = 0; kk < 64; kk += 32) {
            // B fragments (32x16): lane = N (lm); elem e -> K = 16*(e>>3) + 8*lh + (e&7)
            v16bf bfrag[2];
            #pragma unroll
            for (int g = 0; g < 2; ++g) {
                const bf16* p = &Bs[buf][waveN * 32 + g * 16 + lm][kk + lh * 8];
                v8bf b0 = *(const v8bf*)p;
                v8bf b1 = *(const v8bf*)(p + 16);
                v16bf t;
                #pragma unroll
                for (int e = 0; e < 8; ++e) { t[e] = b0[e]; t[8 + e] = b1[e]; }
                bfrag[g] = t;
            }
            // A fragments (16x32): lane = M (lm); elem e -> K = 16*(e>>3) + 8*lh + (e&7)
            #pragma unroll
            for (int f = 0; f < 4; ++f) {
                const bf16* p = &As[buf][waveM * 64 + f * 16 + lm][kk + lh * 8];
                v8bf a0 = *(const v8bf*)p;          // K = 8*lh .. +7
                v8bf a1 = *(const v8bf*)(p + 16);   // K = 16+8*lh .. +7
                v16bf a;
                #pragma unroll
                for (int e = 0; e < 8; ++e) { a[e] = a0[e]; a[8 + e] = a1[e]; }
                #pragma unroll
                for (int g = 0; g < 2; ++g)
                    acc[f][g] = __builtin_amdgcn_wmma_f32_16x16x32_bf16(
                        false, a, false, bfrag[g], (short)0, acc[f][g], false, false);
            }
        }
        buf ^= 1;
    }

    // Epilogue. D layout: lane -> n = lm; rows m = r + 8*lh, value acc[..][r]
    #pragma unroll
    for (int f = 0; f < 4; ++f) {
        const int mBase = mTile + waveM * 64 + f * 16 + lh * 8;
        #pragma unroll
        for (int g = 0; g < 2; ++g) {
            const int nLoc = nTile + waveN * 32 + g * 16 + lm;
            #pragma unroll
            for (int r = 0; r < 8; ++r) {
                float v = acc[f][g][r];
                int   b = mBase + r;
                if (MODE == 0) {
                    float zv = 1.f / (1.f + __expf(-(v + bias[mod * kMS + nLoc])));
                    outF[(size_t)b * kH + mod * kMS + nLoc] = zv;
                } else if (MODE == 1) {
                    float rv = 1.f / (1.f + __expf(-(v + bias[nLoc])));
                    int   d  = c_delay[mod * 4 + (nLoc >> 9)];
                    float hd = (float)Hd[(size_t)d * kB * kH + (size_t)b * kH + nLoc];
                    outB[(size_t)mod * kB * kH + (size_t)b * kH + nLoc] = (bf16)(rv * hd);
                } else if (MODE == 2) {
                    float ht = tanhf(v + bias[mod * kMS + nLoc]);
                    float zv = Zf[(size_t)b * kH + mod * kMS + nLoc];
                    float hp = h_prev[(size_t)b * kH + mod * kMS + nLoc]; // diag delay == 0
                    float hn = (1.f - zv) * hp + zv * ht;
                    outF[(size_t)b * kH + mod * kMS + nLoc] = hn;
                    outB[(size_t)b * kH + mod * kMS + nLoc] = (bf16)hn;
                } else {
                    if (nLoc < kO) {
                        float yv = 1.f / (1.f + __expf(-(v + bias[nLoc])));
                        outF[(size_t)b * kO + nLoc] = yv;
                    }
                }
            }
        }
    }
}

// ---------------- launch ----------------

extern "C" void kernel_launch(void* const* d_in, const int* in_sizes, int n_in,
                              void* d_out, int out_size, void* d_ws, size_t ws_size,
                              hipStream_t stream) {
    (void)in_sizes; (void)n_in; (void)out_size; (void)ws_size;
    const float* x        = (const float*)d_in[0];
    const float* h_prev   = (const float*)d_in[1];
    const float* h_buffer = (const float*)d_in[2];
    const float* Wz = (const float*)d_in[3];
    const float* bz = (const float*)d_in[4];
    const float* Wr = (const float*)d_in[5];
    const float* br = (const float*)d_in[6];
    const float* Wh = (const float*)d_in[7];
    const float* bh = (const float*)d_in[8];
    const float* Y  = (const float*)d_in[9];
    const float* bY = (const float*)d_in[10];

    char*  ws  = (char*)d_ws;
    size_t off = 0;
    auto alloc = [&](size_t bytes) -> void* {
        void* p = ws + off; off += (bytes + 255) & ~(size_t)255; return p;
    };
    bf16*  Wz_bf = (bf16*)alloc((size_t)kH * kKA * 2);
    bf16*  Wr_bf = (bf16*)alloc((size_t)kH * kKA * 2);
    bf16*  Wh_bf = (bf16*)alloc((size_t)kH * kKA * 2);
    bf16*  Yp    = (bf16*)alloc((size_t)128 * kH * 2);
    bf16*  Xbf   = (bf16*)alloc((size_t)kB * kI * 2);
    bf16*  Hd    = (bf16*)alloc((size_t)3 * kB * kH * 2);
    bf16*  Rh    = (bf16*)alloc((size_t)4 * kB * kH * 2);
    float* Zf    = (float*)alloc((size_t)kB * kH * 4);
    bf16*  Hn    = (bf16*)alloc((size_t)kB * kH * 2);

    float* y_out = (float*)d_out;
    float* h_out = y_out + (size_t)kB * kO;

    const int nW = kH * kKA;
    k_f32_to_bf16<<<(nW + 255) / 256, 256, 0, stream>>>(Wz, Wz_bf, nW);
    k_f32_to_bf16<<<(nW + 255) / 256, 256, 0, stream>>>(Wr, Wr_bf, nW);
    k_f32_to_bf16<<<(nW + 255) / 256, 256, 0, stream>>>(Wh, Wh_bf, nW);
    k_prep_y<<<(128 * kH + 255) / 256, 256, 0, stream>>>(Y, Yp);
    k_prep_inputs<<<(kB * kH + 255) / 256, 256, 0, stream>>>(x, h_prev, h_buffer, Xbf, Hd);

    dim3 blk(256);
    gemm_wmma<0><<<dim3(kMS / 128, kB / 128, 4), blk, 0, stream>>>(
        Xbf, Hd, Rh, Hn, Wz_bf, kKA, kKA, bz, nullptr, nullptr, Zf, nullptr);
    gemm_wmma<1><<<dim3(kH / 128, kB / 128, 4), blk, 0, stream>>>(
        Xbf, Hd, Rh, Hn, Wr_bf, kKA, kKA, br, nullptr, nullptr, nullptr, Rh);
    gemm_wmma<2><<<dim3(kMS / 128, kB / 128, 4), blk, 0, stream>>>(
        Xbf, Hd, Rh, Hn, Wh_bf, kKA, kKA, bh, Zf, h_prev, h_out, Hn);
    gemm_wmma<3><<<dim3(1, kB / 128, 1), blk, 0, stream>>>(
        Xbf, Hd, Rh, Hn, Yp, kH, kH, bY, nullptr, nullptr, y_out, nullptr);
}
